// PerlinAttention_45337674777164
// MI455X (gfx1250) — matrix operand order repaired and checked by reading, not compiled, not run on hardware
//
#include <hip/hip_runtime.h>
#include <math.h>

// ---------------- problem constants (B=1) ----------------
#define Hh   12
#define Tt   2048
#define Dd   64
#define Ll   128
#define Mm   266
#define Mp   288          // M padded to a multiple of 32 (WMMA K granularity)
#define VHh  192
#define E2D  128
#define DEC  256
#define C4   48
#define HT   (Hh*Tt)

typedef __bf16 bf16_t;
typedef bf16_t v16bf __attribute__((ext_vector_type(16)));
typedef float  v8f   __attribute__((ext_vector_type(8)));
typedef unsigned short us8 __attribute__((ext_vector_type(8)));

union Frag16 { v16bf v; us8 u[2]; };

#define WMMA_BF16(a, b, c) \
  __builtin_amdgcn_wmma_f32_16x16x32_bf16(false, (a), false, (b), (short)0, (c), false, false)

// ---------------- workspace layout (bytes) ----------------
static constexpr size_t OFF_QN    = 0;                                    // bf16 [H][T][64]  q * 64^-0.25
static constexpr size_t SZ_QN     = (size_t)HT * 64 * 2;
static constexpr size_t OFF_KN    = OFF_QN + SZ_QN;                       // bf16 [H][T][64]
static constexpr size_t OFF_PROJ  = OFF_KN + SZ_QN;                       // bf16 [288][64] (rows >=266 zero)
static constexpr size_t OFF_XPQ   = OFF_PROJ + (size_t)Mp * 64 * 2;       // f32 [H][T][288]; later reused: tperf f32 [H][T][192]
static constexpr size_t SZ_XP     = (size_t)HT * Mp * 4;
static constexpr size_t OFF_XPK   = OFF_XPQ + SZ_XP;                      // f32 [H][T][288]; later reused: dec_out f32 [H][T][256]
static constexpr size_t OFF_DIAGQ = OFF_XPK + SZ_XP;                      // f32 [H*T]
static constexpr size_t OFF_DIAGK = OFF_DIAGQ + (size_t)HT * 4;
static constexpr size_t OFF_RMAX  = OFF_DIAGK + (size_t)HT * 4;
static constexpr size_t OFF_KSTAB = OFF_RMAX + (size_t)HT * 4;            // f32 [H]
static constexpr size_t OFF_PQ    = OFF_KSTAB + 64;                       // bf16 [H][T][288]; later reused: conv out1 f32
static constexpr size_t SZ_PQ     = (size_t)HT * Mp * 2;
static constexpr size_t OFF_PKT   = OFF_PQ + SZ_PQ;                       // bf16 [H][288][T]; later reused: conv out2 f32
static constexpr size_t OFF_PKSUM = OFF_PKT + SZ_PQ;                      // f32 [H][288]
static constexpr size_t OFF_VT    = OFF_PKSUM + (size_t)Hh * Mp * 4;      // bf16 [H][192][T]; later reused: conv out3 f32
static constexpr size_t OFF_KVT   = OFF_VT + (size_t)Hh * VHh * Tt * 2;   // bf16 [H][192][288]  (kv^T)
static constexpr size_t OFF_TPB   = OFF_KVT + (size_t)Hh * VHh * Mp * 2;  // bf16 [H][T][192]  z-scaled t_perf
static constexpr size_t OFF_ENCWT = OFF_TPB + (size_t)HT * VHh * 2;       // bf16 [128][192]  enc_w^T
static constexpr size_t OFF_ENCR  = OFF_ENCWT + (size_t)E2D * VHh * 2;    // f32 [H][T][128]
static constexpr size_t OFF_TENC  = OFF_ENCR + (size_t)HT * E2D * 4;      // bf16 [H][T][128]
static constexpr size_t OFF_DECWT = OFF_TENC + (size_t)HT * E2D * 2;      // bf16 [256][128]  dec_w^T
static constexpr size_t OFF_MASK  = OFF_DECWT + (size_t)DEC * E2D * 2;    // u32 [H][T][4]  128 mask bits / row
static constexpr size_t OFF_VT2   = OFF_MASK + (size_t)HT * 4 * 4;        // bf16 [H][64][T]  v^T
static constexpr size_t WS_TOTAL  = OFF_VT2 + (size_t)HT * 64 * 2;        // ~134 MB

// ---------------- fragment helpers (layouts per CDNA5 ISA) ----------------
// A 16x32 bf16: lane (l&15)=row; lanes<16 hold K {0..7,16..23}, lanes>=16 hold K {8..15,24..31}
__device__ __forceinline__ v16bf load_a_frag(const bf16_t* p, int ld, int lane) {
  const bf16_t* q = p + (long)(lane & 15) * ld + ((lane >> 4) * 8);
  Frag16 f;
  f.u[0] = *(const us8*)(q);
  f.u[1] = *(const us8*)(q + 16);
  return f.v;
}
// B 32x16 bf16 staged column-major (Bt[n][k]): lane (l&15)=col; lanes<16 hold K {0..7,16..23}, lanes>=16 hold K {8..15,24..31}
__device__ __forceinline__ v16bf load_bt_frag(const bf16_t* p, int ld, int lane) {
  const bf16_t* q = p + (long)(lane & 15) * ld + ((lane >> 4) * 8);
  Frag16 f;
  f.u[0] = *(const us8*)(q);
  f.u[1] = *(const us8*)(q + 16);
  return f.v;
}

// ---------------- generic batched GEMM: C[MrxN] = A[MrxK] * Bt[NxK]^T (+bias) ----------------
__global__ __launch_bounds__(128) void gemm_bf16_k(
    const bf16_t* __restrict__ A, long sA,
    const bf16_t* __restrict__ Bt, long sB,
    void* __restrict__ Cv, long sC,
    const float* __restrict__ bias,
    int Mr, int N, int K, int lda, int ldb, int ldc, int transC)
{
  int h = blockIdx.y;
  int lane = threadIdx.x & 31;
  int tile = blockIdx.x * (blockDim.x >> 5) + (threadIdx.x >> 5);
  int nt = N >> 4;
  if (tile >= (Mr >> 4) * nt) return;
  int tm = tile / nt, tn = tile - tm * nt;
  const bf16_t* Ab = A + (long)h * sA + (long)(tm * 16) * lda;
  const bf16_t* Bb = Bt + (long)h * sB + (long)(tn * 16) * ldb;
  v8f acc = {0.f, 0.f, 0.f, 0.f, 0.f, 0.f, 0.f, 0.f};
  for (int k = 0; k < K; k += 32) {
    v16bf a = load_a_frag(Ab + k, lda, lane);
    v16bf b = load_bt_frag(Bb + k, ldb, lane);
    acc = WMMA_BF16(a, b, acc);
  }
  int col = lane & 15, rb = (lane >> 4) * 8;
  if (!transC) {
    float* C = (float*)Cv + (long)h * sC;
    float bv = bias ? bias[tn * 16 + col] : 0.f;
#pragma unroll
    for (int r = 0; r < 8; ++r)
      C[(long)(tm * 16 + rb + r) * ldc + tn * 16 + col] = acc[r] + bv;
  } else {  // store D^T directly in bf16 (builds staged Bt for the next GEMM)
    bf16_t* C = (bf16_t*)Cv + (long)h * sC;
#pragma unroll
    for (int r = 0; r < 8; ++r)
      C[(long)(tn * 16 + col) * ldc + tm * 16 + rb + r] = (bf16_t)acc[r];
  }
}

// ---------------- prep kernels ----------------
// qn = q * 64^-0.25 (bf16) + diag = 0.5*sum(qn^2); one wave per row of 64
__global__ __launch_bounds__(256) void prep_x_k(const float* __restrict__ x,
                                                bf16_t* __restrict__ xn, float* __restrict__ diag)
{
  int lane = threadIdx.x & 31;
  long row = (long)blockIdx.x * 8 + (threadIdx.x >> 5);
  const float* xr = x + row * Dd;
  const float c = 0.35355339059327373f;  // 64^-0.25
  float a = xr[lane] * c;
  float b = xr[lane + 32] * c;
  xn[row * Dd + lane] = (bf16_t)a;
  xn[row * Dd + lane + 32] = (bf16_t)b;
  float ss = a * a + b * b;
  for (int o = 1; o < 32; o <<= 1) ss += __shfl_xor(ss, o, 32);
  if (lane == 0) diag[row] = 0.5f * ss;
}

__global__ void prep_proj_k(const float* __restrict__ proj, bf16_t* __restrict__ pp) {
  int idx = blockIdx.x * blockDim.x + threadIdx.x;
  if (idx >= Mp * Dd) return;
  pp[idx] = (bf16_t)((idx < Mm * Dd) ? proj[idx] : 0.f);  // zero-pad rows 266..287
}

// batched [R][C] -> bf16 [C][R]
__global__ void transpose_bf16_k(const float* __restrict__ in, bf16_t* __restrict__ out, int R, int C) {
  long base = (long)blockIdx.y * R * C;
  int n = R * C;
  for (int idx = blockIdx.x * blockDim.x + threadIdx.x; idx < n; idx += gridDim.x * blockDim.x) {
    int r = idx / C, c = idx - r * C;
    out[base + (long)c * R + r] = (bf16_t)in[base + idx];
  }
}

// ---------------- performer feature kernels ----------------
__global__ __launch_bounds__(256) void rowmax_k(const float* __restrict__ xp, float* __restrict__ rmax) {
  int lane = threadIdx.x & 31;
  long row = (long)blockIdx.x * 8 + (threadIdx.x >> 5);
  const float* xr = xp + row * Mp;
  float mx = -3.4e38f;
  for (int m = lane; m < Mm; m += 32) mx = fmaxf(mx, xr[m]);
  for (int o = 1; o < 32; o <<= 1) mx = fmaxf(mx, __shfl_xor(mx, o, 32));
  if (lane == 0) rmax[row] = mx;
}

__global__ __launch_bounds__(256) void headmax_k(const float* __restrict__ rmax, float* __restrict__ kstab) {
  __shared__ float red[256];
  int h = blockIdx.x, tid = threadIdx.x;
  float mx = -3.4e38f;
  for (int i = tid; i < Tt; i += 256) mx = fmaxf(mx, rmax[(long)h * Tt + i]);
  red[tid] = mx;
  __syncthreads();
  for (int s = 128; s > 0; s >>= 1) {
    if (tid < s) red[tid] = fmaxf(red[tid], red[tid + s]);
    __syncthreads();
  }
  if (tid == 0) kstab[h] = red[0];
}

__global__ __launch_bounds__(256) void expq_k(const float* __restrict__ xp, const float* __restrict__ diag,
                                              bf16_t* __restrict__ pq)
{
  int lane = threadIdx.x & 31;
  long row = (long)blockIdx.x * 8 + (threadIdx.x >> 5);
  const float* xr = xp + row * Mp;
  float dg = diag[row];
  float mx = -3.4e38f;
  for (int m = lane; m < Mm; m += 32) mx = fmaxf(mx, xr[m]);
  for (int o = 1; o < 32; o <<= 1) mx = fmaxf(mx, __shfl_xor(mx, o, 32));
  const float cm = 0.061313933f;  // 266^-0.5
  for (int m = lane; m < Mp; m += 32) {
    float v = (m < Mm) ? cm * (__expf(xr[m] - dg - mx) + 1e-6f) : 0.f;
    pq[row * Mp + m] = (bf16_t)v;
  }
}

__global__ void expk_k(const float* __restrict__ xp, const float* __restrict__ diag,
                       const float* __restrict__ kstab, bf16_t* __restrict__ pkT)
{
  long idx = (long)blockIdx.x * blockDim.x + threadIdx.x;
  if (idx >= (long)HT * Mm) return;
  int m = (int)(idx % Mm);
  long row = idx / Mm;  // h*T + t
  int h = (int)(row / Tt);
  int t = (int)(row - (long)h * Tt);
  float v = 0.061313933f * (__expf(xp[row * Mp + m] - diag[row] - kstab[h]) + 1e-6f);
  pkT[((long)h * Mp + m) * Tt + t] = (bf16_t)v;  // transposed store -> pk^T is GEMM-A next
}

__global__ void padpkT_k(bf16_t* __restrict__ pkT) {
  int idx = blockIdx.x * blockDim.x + threadIdx.x;  // 12*22*2048 exactly
  int t = idx & (Tt - 1);
  int mm = (idx >> 11) % 22;
  int h = idx / (22 * Tt);
  pkT[((long)h * Mp + Mm + mm) * Tt + t] = (bf16_t)0.f;
}

__global__ __launch_bounds__(256) void pksum_k(const bf16_t* __restrict__ pkT, float* __restrict__ ps) {
  int lane = threadIdx.x & 31;
  long row = (long)blockIdx.x * 8 + (threadIdx.x >> 5);  // h*288 + m
  const bf16_t* p = pkT + row * Tt;
  float s = 0.f;
  for (int i = lane; i < Tt; i += 32) s += (float)p[i];
  for (int o = 1; o < 32; o <<= 1) s += __shfl_xor(s, o, 32);
  if (lane == 0) ps[row] = s;
}

// z = 1/(pq . pksum + 1e-8); t_perf *= z; convert to bf16
__global__ __launch_bounds__(256) void zscale_k(const bf16_t* __restrict__ pq, const float* __restrict__ ps,
                                                const float* __restrict__ tp, bf16_t* __restrict__ tpb)
{
  int lane = threadIdx.x & 31;
  long row = (long)blockIdx.x * 8 + (threadIdx.x >> 5);  // h*T + t
  int h = (int)(row / Tt);
  const bf16_t* pr = pq + row * Mp;
  const float* ph = ps + (long)h * Mp;
  float s = 0.f;
  for (int m = lane; m < Mp; m += 32) s += (float)pr[m] * ph[m];
  for (int o = 1; o < 32; o <<= 1) s += __shfl_xor(s, o, 32);
  float z = 1.f / (s + 1e-8f);
  for (int d = lane; d < VHh; d += 32)
    tpb[row * VHh + d] = (bf16_t)(tp[row * VHh + d] * z);
}

// LayerNorm (population var) + exact GELU, one wave per 128-wide row
__global__ __launch_bounds__(256) void lngelu_k(const float* __restrict__ x, const float* __restrict__ g,
                                                const float* __restrict__ b, bf16_t* __restrict__ out)
{
  int lane = threadIdx.x & 31;
  long row = (long)blockIdx.x * 8 + (threadIdx.x >> 5);
  const float* xr = x + row * E2D;
  float v0 = xr[lane], v1 = xr[lane + 32], v2 = xr[lane + 64], v3 = xr[lane + 96];
  float s = v0 + v1 + v2 + v3;
  float q = v0 * v0 + v1 * v1 + v2 * v2 + v3 * v3;
  for (int o = 1; o < 32; o <<= 1) { s += __shfl_xor(s, o, 32); q += __shfl_xor(q, o, 32); }
  float mu = s * (1.f / 128.f);
  float var = q * (1.f / 128.f) - mu * mu;
  float rs = rsqrtf(var + 1e-5f);
  float vals[4] = {v0, v1, v2, v3};
#pragma unroll
  for (int i = 0; i < 4; ++i) {
    int c = lane + 32 * i;
    float y = (vals[i] - mu) * rs * g[c] + b[c];
    float ge = 0.5f * y * (1.f + erff(y * 0.70710678f));
    out[row * E2D + c] = (bf16_t)ge;
  }
}

// ---------------- CNN predictor (direct conv, fp32 VALU) ----------------
// conv1: in = dec_out [H][T][256] viewed as channels ch=h*4+c -> slice (c*64+l); stride (2,1), pad 1, ReLU
__global__ __launch_bounds__(256) void conv1_k(const float* __restrict__ x, const float* __restrict__ w,
                                               const float* __restrict__ b, float* __restrict__ y)
{
  int idx = blockIdx.x * blockDim.x + threadIdx.x;  // [48][1024][64]
  int ox = idx & 63;
  int oy = (idx >> 6) & 1023;
  int oc = idx >> 16;
  float acc = b[oc];
  for (int ic = 0; ic < C4; ++ic) {
    const float* xin = x + ((long)(ic >> 2) * Tt) * DEC + (ic & 3) * 64;
    const float* wp = w + ((long)oc * C4 + ic) * 9;
#pragma unroll
    for (int ky = 0; ky < 3; ++ky) {
      int iy = oy * 2 + ky - 1;
      if ((unsigned)iy >= (unsigned)Tt) continue;
#pragma unroll
      for (int kx = 0; kx < 3; ++kx) {
        int ix = ox + kx - 1;
        if ((unsigned)ix >= 64u) continue;
        acc += xin[(long)iy * DEC + ix] * wp[ky * 3 + kx];
      }
    }
  }
  y[idx] = fmaxf(acc, 0.f);
}

__global__ __launch_bounds__(256) void conv2_k(const float* __restrict__ x, const float* __restrict__ w,
                                               const float* __restrict__ b, float* __restrict__ y)
{
  int idx = blockIdx.x * blockDim.x + threadIdx.x;  // [48][1024][64]
  int ox = idx & 63;
  int oy = (idx >> 6) & 1023;
  int oc = idx >> 16;
  float acc = b[oc];
  for (int ic = 0; ic < C4; ++ic) {
    const float* xin = x + ((long)ic * 1024) * 64;
    const float* wp = w + ((long)oc * C4 + ic) * 9;
#pragma unroll
    for (int ky = 0; ky < 3; ++ky) {
      int iy = oy + ky - 1;
      if ((unsigned)iy >= 1024u) continue;
#pragma unroll
      for (int kx = 0; kx < 3; ++kx) {
        int ix = ox + kx - 1;
        if ((unsigned)ix >= 64u) continue;
        acc += xin[(long)iy * 64 + ix] * wp[ky * 3 + kx];
      }
    }
  }
  y[idx] = fmaxf(acc, 0.f);
}

// conv3 on nearest-upsampled input (row iy of upsampled = out2 row iy>>1); no ReLU
__global__ __launch_bounds__(256) void conv3_k(const float* __restrict__ x, const float* __restrict__ w,
                                               const float* __restrict__ b, float* __restrict__ y)
{
  int idx = blockIdx.x * blockDim.x + threadIdx.x;  // [12][2048][64]
  int ox = idx & 63;
  int oy = (idx >> 6) & 2047;
  int oc = idx >> 17;
  float acc = b[oc];
  for (int ic = 0; ic < C4; ++ic) {
    const float* xin = x + ((long)ic * 1024) * 64;
    const float* wp = w + ((long)oc * C4 + ic) * 9;
#pragma unroll
    for (int ky = 0; ky < 3; ++ky) {
      int iy = oy + ky - 1;
      if ((unsigned)iy >= (unsigned)Tt) continue;
      int iy2 = iy >> 1;
#pragma unroll
      for (int kx = 0; kx < 3; ++kx) {
        int ix = ox + kx - 1;
        if ((unsigned)ix >= 64u) continue;
        acc += xin[(long)iy2 * 64 + ix] * wp[ky * 3 + kx];
      }
    }
  }
  y[idx] = acc;
}

// ---------------- mask: rank-select threshold + 128-bit mask per query row ----------------
// est_full[s] = softmax(dup(x))[s>>4]; softmax is monotone and every raw value appears
// (T/L)*2 = 32 times in est_full, so "top-k over 2048" reduces to a small rank over the
// 64 raw conv outputs: rank64 = ceil(ceil(k/16)/2). mask bit per L-cell l: x[l>>1] >= th.
__global__ __launch_bounds__(256) void mask_k(const float* __restrict__ x3, const int* __restrict__ ktopk,
                                              unsigned* __restrict__ mb)
{
  __shared__ float xs[8][64];
  int lane = threadIdx.x & 31;
  int wave = threadIdx.x >> 5;
  long row = (long)blockIdx.x * 8 + wave;  // h*T + t
  const float* xr = x3 + row * 64;
  float a = xr[lane], b = xr[lane + 32];
  xs[wave][lane] = a;
  xs[wave][lane + 32] = b;
  asm volatile("s_wait_dscnt 0x0" ::: "memory");  // wave-private LDS, in-wave RAW
  int kk = ktopk[0];
  int rank128 = (kk + 15) >> 4;
  if (rank128 < 1) rank128 = 1;
  int rank = (rank128 + 1) >> 1;  // rank within the 64 raw values
  int ca = 0, cb = 0;
  for (int i = 0; i < 64; ++i) {
    float t = xs[wave][i];
    ca += (t >= a);
    cb += (t >= b);
  }
  float cand = fmaxf(ca >= rank ? a : -3.4e38f, cb >= rank ? b : -3.4e38f);
  for (int o = 1; o < 32; o <<= 1) cand = fmaxf(cand, __shfl_xor(cand, o, 32));
  float th = cand;
#pragma unroll
  for (int w = 0; w < 4; ++w) {
    bool pred = xs[wave][(w * 32 + lane) >> 1] >= th;
    unsigned long long bal = __ballot(pred);
    if (lane == 0) mb[row * 4 + w] = (unsigned)bal;
  }
}

// ---------------- fused masked flash attention ----------------
// scores = qn.kn = q.k * 64^-0.5 (the performer prescale gives the 1/sqrt(D) for free).
// 8 waves/block, each wave owns a 16-row query tile, streams 32 keys/step.
__global__ __launch_bounds__(256) void flash_attn_k(
    const bf16_t* __restrict__ qn, const bf16_t* __restrict__ kn,
    const bf16_t* __restrict__ vT, const unsigned* __restrict__ mb,
    float* __restrict__ out)
{
  __shared__ bf16_t plds[8][16][36];  // P tile bounce buffer (C-layout -> A-layout)
  int lane = threadIdx.x & 31;
  int wave = threadIdx.x >> 5;
  int h = blockIdx.y;
  int t0 = (blockIdx.x * 8 + wave) * 16;
  const bf16_t* Q = qn + ((long)h * Tt + t0) * Dd;
  const bf16_t* Kp = kn + (long)h * Tt * Dd;
  const bf16_t* V = vT + (long)h * Dd * Tt;
  int arow = lane & 15, akb = (lane >> 4) * 8;
  int col = lane & 15, kb16 = (lane >> 4) * 16, rb = (lane >> 4) * 8;

  Frag16 aq0, aq1;
  {
    const bf16_t* p = Q + (long)arow * Dd;
    aq0.u[0] = *(const us8*)(p + akb);
    aq0.u[1] = *(const us8*)(p + akb + 16);
    aq1.u[0] = *(const us8*)(p + 32 + akb);
    aq1.u[1] = *(const us8*)(p + 32 + akb + 16);
  }
  // lane i<16 keeps the 4 mask words of query row i (shuffled out per step)
  const unsigned* mrow = mb + ((long)h * Tt + t0 + (lane & 15)) * 4;
  unsigned mw0 = mrow[0], mw1 = mrow[1], mw2 = mrow[2], mw3 = mrow[3];

  const v8f vzero = {0.f, 0.f, 0.f, 0.f, 0.f, 0.f, 0.f, 0.f};
  v8f acc[4];
#pragma unroll
  for (int j = 0; j < 4; ++j) acc[j] = vzero;
  float mrun[8], lrun[8];
#pragma unroll
  for (int r = 0; r < 8; ++r) { mrun[r] = -1e30f; lrun[r] = 0.f; }

  for (int s0 = 0; s0 < Tt; s0 += 32) {
    v8f sA = vzero, sB = vzero;
    {
      const bf16_t* p0 = Kp + (long)(s0 + col) * Dd + akb;
      const bf16_t* p1 = Kp + (long)(s0 + 16 + col) * Dd + akb;
      Frag16 b0, b1, b2, b3;
      b0.u[0] = *(const us8*)(p0);      b0.u[1] = *(const us8*)(p0 + 16);
      b1.u[0] = *(const us8*)(p0 + 32); b1.u[1] = *(const us8*)(p0 + 48);
      b2.u[0] = *(const us8*)(p1);      b2.u[1] = *(const us8*)(p1 + 16);
      b3.u[0] = *(const us8*)(p1 + 32); b3.u[1] = *(const us8*)(p1 + 48);
      sA = WMMA_BF16(aq0.v, b0.v, sA);
      sA = WMMA_BF16(aq1.v, b1.v, sA);
      sB = WMMA_BF16(aq0.v, b2.v, sB);
      sB = WMMA_BF16(aq1.v, b3.v, sB);
    }
    int cell0 = s0 >> 4;           // two L-cells per 32-key step
    int wsel = cell0 >> 5;
    int bpos = cell0 & 31;
    unsigned msel = (wsel == 0) ? mw0 : (wsel == 1) ? mw1 : (wsel == 2) ? mw2 : mw3;
#pragma unroll
    for (int r = 0; r < 8; ++r) {
      unsigned wm = __shfl(msel, rb + r, 32);
      float v0 = ((wm >> bpos) & 1u) ? sA[r] : -1e9f;
      float v1 = ((wm >> (bpos + 1)) & 1u) ? sB[r] : -1e9f;
      float rx = fmaxf(v0, v1);
      rx = fmaxf(rx, __shfl_xor(rx, 1, 32));
      rx = fmaxf(rx, __shfl_xor(rx, 2, 32));
      rx = fmaxf(rx, __shfl_xor(rx, 4, 32));
      rx = fmaxf(rx, __shfl_xor(rx, 8, 32));   // 16-lane row reduce (rows 0-7 | 8-15)
      float mn = fmaxf(mrun[r], rx);
      float sc = __expf(mrun[r] - mn);
      float e0 = __expf(v0 - mn);
      float e1 = __expf(v1 - mn);
      float rs = e0 + e1;
      rs += __shfl_xor(rs, 1, 32);
      rs += __shfl_xor(rs, 2, 32);
      rs += __shfl_xor(rs, 4, 32);
      rs += __shfl_xor(rs, 8, 32);
      lrun[r] = lrun[r] * sc + rs;
      mrun[r] = mn;
      acc[0][r] *= sc; acc[1][r] *= sc; acc[2][r] *= sc; acc[3][r] *= sc;
      plds[wave][rb + r][col] = (bf16_t)e0;
      plds[wave][rb + r][16 + col] = (bf16_t)e1;
    }
    asm volatile("s_wait_dscnt 0x0" ::: "memory");  // wave-private P tile ready
    Frag16 ap;
#pragma unroll
    for (int i = 0; i < 8; ++i) {
      ap.v[i] = plds[wave][arow][akb + i];
      ap.v[8 + i] = plds[wave][arow][akb + 16 + i];
    }
#pragma unroll
    for (int j = 0; j < 4; ++j) {
      const bf16_t* vp = V + (long)(j * 16 + col) * Tt + s0 + akb;
      Frag16 bv;
      bv.u[0] = *(const us8*)(vp);
      bv.u[1] = *(const us8*)(vp + 16);
      acc[j] = WMMA_BF16(ap.v, bv.v, acc[j]);
    }
  }
#pragma unroll
  for (int r = 0; r < 8; ++r) {
    float inv = 1.f / lrun[r];
    long o = ((long)h * Tt + t0 + rb + r) * Dd + col;
    out[o] = acc[0][r] * inv;
    out[o + 16] = acc[1][r] * inv;
    out[o + 32] = acc[2][r] * inv;
    out[o + 48] = acc[3][r] * inv;
  }
}

// ---------------- orchestration ----------------
extern "C" void kernel_launch(void* const* d_in, const int* in_sizes, int n_in,
                              void* d_out, int out_size, void* d_ws, size_t ws_size,
                              hipStream_t stream)
{
  (void)in_sizes; (void)n_in; (void)out_size;
  if (ws_size < WS_TOTAL) return;

  const float* q     = (const float*)d_in[0];
  const float* k     = (const float*)d_in[1];
  const float* v     = (const float*)d_in[2];
  const float* vfa   = (const float*)d_in[3];
  const float* proj  = (const float*)d_in[4];
  const float* enc_w = (const float*)d_in[5];
  const float* enc_b = (const float*)d_in[6];
  const float* ln_g  = (const float*)d_in[7];
  const float* ln_b  = (const float*)d_in[8];
  const float* dec_w = (const float*)d_in[9];
  const float* dec_b = (const float*)d_in[10];
  const float* c1w   = (const float*)d_in[11];
  const float* c1b   = (const float*)d_in[12];
  const float* c2w   = (const float*)d_in[13];
  const float* c2b   = (const float*)d_in[14];
  const float* c3w   = (const float*)d_in[15];
  const float* c3b   = (const float*)d_in[16];
  const int*   ktop  = (const int*)d_in[17];
  float* out = (float*)d_out;
  char* ws = (char*)d_ws;

  bf16_t* qn    = (bf16_t*)(ws + OFF_QN);
  bf16_t* kn    = (bf16_t*)(ws + OFF_KN);
  bf16_t* projp = (bf16_t*)(ws + OFF_PROJ);
  float*  xpq   = (float*)(ws + OFF_XPQ);
  float*  tperf = (float*)(ws + OFF_XPQ);   // alias: xpq dead after expq
  float*  xpk   = (float*)(ws + OFF_XPK);
  float*  decb  = (float*)(ws + OFF_XPK);   // alias: xpk dead after expk
  float*  diagq = (float*)(ws + OFF_DIAGQ);
  float*  diagk = (float*)(ws + OFF_DIAGK);
  float*  rmax  = (float*)(ws + OFF_RMAX);
  float*  kstab = (float*)(ws + OFF_KSTAB);
  bf16_t* pq    = (bf16_t*)(ws + OFF_PQ);
  float*  out1  = (float*)(ws + OFF_PQ);    // alias: pq dead after t_perf/z
  bf16_t* pkT   = (bf16_t*)(ws + OFF_PKT);
  float*  out2  = (float*)(ws + OFF_PKT);   // alias: pkT dead after kv/pksum
  float*  pks   = (float*)(ws + OFF_PKSUM);
  bf16_t* vT    = (bf16_t*)(ws + OFF_VT);
  float*  out3  = (float*)(ws + OFF_VT);    // alias: vT dead after kv GEMM
  bf16_t* kvT   = (bf16_t*)(ws + OFF_KVT);
  bf16_t* tpb   = (bf16_t*)(ws + OFF_TPB);
  bf16_t* encwT = (bf16_t*)(ws + OFF_ENCWT);
  float*  encr  = (float*)(ws + OFF_ENCR);
  bf16_t* tenc  = (bf16_t*)(ws + OFF_TENC);
  bf16_t* decwT = (bf16_t*)(ws + OFF_DECWT);
  unsigned* maskb = (unsigned*)(ws + OFF_MASK);
  bf16_t* vT2   = (bf16_t*)(ws + OFF_VT2);

  // --- prep / conversions ---
  prep_x_k<<<HT / 8, 256, 0, stream>>>(q, qn, diagq);
  prep_x_k<<<HT / 8, 256, 0, stream>>>(k, kn, diagk);
  prep_proj_k<<<(Mp * Dd + 255) / 256, 256, 0, stream>>>(proj, projp);
  transpose_bf16_k<<<dim3(1536, Hh), 256, 0, stream>>>(vfa, vT, Tt, VHh);
  transpose_bf16_k<<<dim3(512, Hh), 256, 0, stream>>>(v, vT2, Tt, Dd);
  transpose_bf16_k<<<dim3(96, 1), 256, 0, stream>>>(enc_w, encwT, VHh, E2D);
  transpose_bf16_k<<<dim3(128, 1), 256, 0, stream>>>(dec_w, decwT, E2D, DEC);

  // --- performer: projections (xp = xn @ proj^T), K=64 ---
  gemm_bf16_k<<<dim3(576, Hh), 128, 0, stream>>>(qn, (long)Tt * Dd, projp, 0, xpq,
                                                 (long)Tt * Mp, nullptr, Tt, Mp, Dd, Dd, Dd, Mp, 0);
  gemm_bf16_k<<<dim3(576, Hh), 128, 0, stream>>>(kn, (long)Tt * Dd, projp, 0, xpk,
                                                 (long)Tt * Mp, nullptr, Tt, Mp, Dd, Dd, Dd, Mp, 0);

  // --- stabilizers + exp features ---
  rowmax_k<<<HT / 8, 256, 0, stream>>>(xpk, rmax);
  headmax_k<<<Hh, 256, 0, stream>>>(rmax, kstab);
  expq_k<<<HT / 8, 256, 0, stream>>>(xpq, diagq, pq);
  expk_k<<<((long)HT * Mm + 255) / 256, 256, 0, stream>>>(xpk, diagk, kstab, pkT);
  padpkT_k<<<(Hh * 22 * Tt) / 256, 256, 0, stream>>>(pkT);
  pksum_k<<<Hh * Mp / 8, 256, 0, stream>>>(pkT, pks);

  // --- kv = pk^T @ v_for_atten (K=2048), stored transposed bf16 for next GEMM ---
  gemm_bf16_k<<<dim3(54, Hh), 128, 0, stream>>>(pkT, (long)Mp * Tt, vT, (long)VHh * Tt, kvT,
                                                (long)VHh * Mp, nullptr, Mp, VHh, Tt, Tt, Tt, Mp, 1);
  // --- t_perf = pq @ kv (K=288 padded) ---
  gemm_bf16_k<<<dim3(384, Hh), 128, 0, stream>>>(pq, (long)Tt * Mp, kvT, (long)VHh * Mp, tperf,
                                                 (long)Tt * VHh, nullptr, Tt, VHh, Mp, Mp, Mp, VHh, 0);
  zscale_k<<<HT / 8, 256, 0, stream>>>(pq, pks, tperf, tpb);

  // --- encoder: Linear(192->128)+bias, LN+GELU ---
  gemm_bf16_k<<<dim3(256, Hh), 128, 0, stream>>>(tpb, (long)Tt * VHh, encwT, 0, encr,
                                                 (long)Tt * E2D, enc_b, Tt, E2D, VHh, VHh, VHh, E2D, 0);
  lngelu_k<<<HT / 8, 256, 0, stream>>>(encr, ln_g, ln_b, tenc);

  // --- decoder: Linear(128->256)+bias (channel-split handled by conv1 indexing) ---
  gemm_bf16_k<<<dim3(512, Hh), 128, 0, stream>>>(tenc, (long)Tt * E2D, decwT, 0, decb,
                                                 (long)Tt * DEC, dec_b, Tt, DEC, E2D, E2D, E2D, DEC, 0);

  // --- CNN predictor ---
  conv1_k<<<(C4 * 1024 * 64) / 256, 256, 0, stream>>>(decb, c1w, c1b, out1);
  conv2_k<<<(C4 * 1024 * 64) / 256, 256, 0, stream>>>(out1, c2w, c2b, out2);
  conv3_k<<<(Hh * Tt * 64) / 256, 256, 0, stream>>>(out2, c3w, c3b, out3);

  // --- top-k mask bits + fused masked flash attention ---
  mask_k<<<HT / 8, 256, 0, stream>>>(out3, ktop, maskb);
  flash_attn_k<<<dim3(Tt / 128, Hh), 256, 0, stream>>>(qn, kn, vT2, maskb, out);
}
